// FBANetLayer_17575006175258
// MI455X (gfx1250) — matrix operand order repaired and checked by reading, not compiled, not run on hardware
//
#include <hip/hip_runtime.h>
#include <hip/hip_bf16.h>
#include <stdint.h>

// ---------------------------------------------------------------------------
// FBANet layer (Swin-style shifted-window attention block) for gfx1250.
// All GEMMs run in bf16 through v_wmma_f32_16x16x32_bf16.
// GEMMs compute D = W_tile x X_tile^T so each lane's 8 accumulator entries are
// 8 consecutive output features of one token -> fully vectorized epilogues.
// ---------------------------------------------------------------------------

typedef __bf16 bf16;
typedef __attribute__((ext_vector_type(16))) bf16  bf16x16;
typedef __attribute__((ext_vector_type(8)))  float f32x8;

#define H_IMG  512
#define W_IMG  512
#define WSZ    8
#define SHIFT  4
#define DIM_   128
#define HEADS_ 4
#define HD_    32
#define HID_   512
#define NTOK   (H_IMG * W_IMG)          // 262144 tokens
#define NWIN   4096                     // (512/8)^2 windows
#define NTOKW  64                       // tokens per window

union Frag { bf16x16 v; uint4 u[2]; };
union BF4  { uint2 u; bf16 e[4]; };
union BF8  { uint4 u; bf16 e[8]; };

__device__ __forceinline__ float gelu_f(float x) {
  return 0.5f * x * (1.0f + tanhf(0.7978845608028654f * (x + 0.044715f * x * x * x)));
}

__device__ __forceinline__ int region_of(int p) {
  // ids: [0,504)->0  [504,508)->1  [508,512)->2
  return (p < H_IMG - WSZ) ? 0 : ((p < H_IMG - SHIFT) ? 1 : 2);
}

__device__ __forceinline__ float wave_sum(float v) {
#pragma unroll
  for (int off = 16; off >= 1; off >>= 1) v += __shfl_xor(v, off, 32);
  return v;
}

// ---------------------------------------------------------------------------
// fp32 -> bf16 weight conversion
// ---------------------------------------------------------------------------
__global__ void f2bf_kernel(const float* __restrict__ in, bf16* __restrict__ out, int n) {
  int i = blockIdx.x * 256 + threadIdx.x;
  if (i < n) out[i] = (bf16)in[i];
}

// ---------------------------------------------------------------------------
// Precompute fused (relative-position bias + shift mask) table:
//   bm[cls][h][n][m], cls = (wr==63)*2 + (wc==63)  (only 4 window classes)
// 65536 fp32 entries = 256KB, stays hot in L2 for the attention pass.
// ---------------------------------------------------------------------------
__global__ void bm_kernel(const float* __restrict__ rel_bias, float* __restrict__ bm) {
  int tid = blockIdx.x * 256 + threadIdx.x;          // [0, 65536)
  int m = tid & 63, n = (tid >> 6) & 63, h = (tid >> 12) & 3, cls = tid >> 14;
  int in_ = n >> 3, jn = n & 7, im = m >> 3, jm = m & 7;
  int idx = (in_ - im + 7) * 15 + (jn - jm + 7);
  float v = rel_bias[idx * HEADS_ + h];
  int wr = (cls & 2) ? 63 : 0, wc = (cls & 1) ? 63 : 0;
  int regn = 3 * region_of(wr * WSZ + in_) + region_of(wc * WSZ + jn);
  int regm = 3 * region_of(wr * WSZ + im) + region_of(wc * WSZ + jm);
  if (regn != regm) v -= 100.0f;
  bm[tid] = v;
}

// ---------------------------------------------------------------------------
// LayerNorm over DIM=128.  SHIFTED=1: gather through the (-4,-4) roll +
// 8x8 window partition, producing xw[NWIN*64][128] in window order.
// One wave per row; 8 waves per block.
// ---------------------------------------------------------------------------
template <int SHIFTED>
__global__ __launch_bounds__(256)
void ln_kernel(const float* __restrict__ in, const float* __restrict__ g,
               const float* __restrict__ b, bf16* __restrict__ out) {
  int wave = threadIdx.x >> 5;
  int lane = threadIdx.x & 31;
  long r = (long)blockIdx.x * 8 + wave;     // output row (window order if SHIFTED)
  long src;
  if (SHIFTED) {
    int w  = (int)(r >> 6), n = (int)(r & 63);
    int wr = w >> 6, wc = w & 63;
    int i  = n >> 3, j = n & 7;
    int gr = (wr * WSZ + i + SHIFT) & (H_IMG - 1);
    int gc = (wc * WSZ + j + SHIFT) & (W_IMG - 1);
    src = (long)gr * W_IMG + gc;
  } else {
    src = r;
  }
  float4 x4 = ((const float4*)(in + src * DIM_))[lane];
  float s = wave_sum(x4.x + x4.y + x4.z + x4.w);
  float mean = s * (1.0f / DIM_);
  float dx0 = x4.x - mean, dx1 = x4.y - mean, dx2 = x4.z - mean, dx3 = x4.w - mean;
  float q = wave_sum(dx0 * dx0 + dx1 * dx1 + dx2 * dx2 + dx3 * dx3);
  float rinv = rsqrtf(q * (1.0f / DIM_) + 1e-5f);
  float4 g4 = ((const float4*)g)[lane];
  float4 b4 = ((const float4*)b)[lane];
  BF4 o;
  o.e[0] = (bf16)(dx0 * rinv * g4.x + b4.x);
  o.e[1] = (bf16)(dx1 * rinv * g4.y + b4.y);
  o.e[2] = (bf16)(dx2 * rinv * g4.z + b4.z);
  o.e[3] = (bf16)(dx3 * rinv * g4.w + b4.w);
  *(uint2*)(out + r * DIM_ + lane * 4) = o.u;
}

// ---------------------------------------------------------------------------
// bf16 WMMA GEMM:  out[t][f] = sum_k X[t][k] * W[f][k]  (+bias, epilogue)
// One wave computes a 64-token x 64-feature tile as D = W x X^T:
//   A-frags from W rows, B-frags from X rows (both 2x b128 loads).
//   Lane (lr, half) of D tile (fi, ti): token = t0+ti*16+lr,
//   features fb..fb+7 with fb = f0+fi*16+8*half  -> vector stores.
// Epilogues:
//   EPI=0: bf16 out, +bias                 (QKV)
//   EPI=1: bf16 out, +bias, GELU           (MLP up)
//   EPI=2: f32 out,  +bias, +res           (MLP down + residual)
//   EPI=3: f32 out,  +bias, window-reverse scatter, +skip (proj)
// ---------------------------------------------------------------------------
template <int EPI>
__global__ __launch_bounds__(32)
void gemm_kernel(const bf16* __restrict__ X, const bf16* __restrict__ Wt,
                 const float* __restrict__ bias, int K, int Nout,
                 bf16* __restrict__ outB, float* __restrict__ outF,
                 const float* __restrict__ res) {
  int lane = threadIdx.x & 31;
  int lr = lane & 15, half = lane >> 4;
  long m0 = (long)blockIdx.x * 64;   // token block
  int f0 = blockIdx.y * 64;          // feature block

  // warm the tail cachelines of each token row we are about to stream
  if (K > 64) {
#pragma unroll
    for (int ti = 0; ti < 4; ++ti)
      __builtin_prefetch(X + (m0 + ti * 16 + lr) * (long)K + 64, 0, 0);
  }

  f32x8 acc[4][4] = {};              // [fi][ti]
  for (int k0 = 0; k0 < K; k0 += 32) {
    Frag af[4], bf[4];
#pragma unroll
    for (int fi = 0; fi < 4; ++fi) {
      const bf16* p = Wt + (long)(f0 + fi * 16 + lr) * K + k0 + half * 8;
      af[fi].u[0] = *(const uint4*)p;
      af[fi].u[1] = *(const uint4*)(p + 16);
    }
#pragma unroll
    for (int ti = 0; ti < 4; ++ti) {
      const bf16* p = X + (m0 + ti * 16 + lr) * (long)K + k0 + half * 8;
      bf[ti].u[0] = *(const uint4*)p;
      bf[ti].u[1] = *(const uint4*)(p + 16);
    }
#pragma unroll
    for (int fi = 0; fi < 4; ++fi)
#pragma unroll
      for (int ti = 0; ti < 4; ++ti)
        acc[fi][ti] = __builtin_amdgcn_wmma_f32_16x16x32_bf16(
            false, af[fi].v, false, bf[ti].v, (short)0, acc[fi][ti], false, false);
  }

#pragma unroll
  for (int ti = 0; ti < 4; ++ti) {
    long t = m0 + ti * 16 + lr;      // token (window order)
    long trow = t;
    if (EPI == 3) {                  // reverse window partition + reverse roll
      int wI = (int)(t >> 6), nI = (int)(t & 63);
      int wr = wI >> 6, wc = wI & 63;
      int i  = nI >> 3, j = nI & 7;
      int gr = (wr * WSZ + i + SHIFT) & (H_IMG - 1);
      int gc = (wc * WSZ + j + SHIFT) & (W_IMG - 1);
      trow = (long)gr * W_IMG + gc;
    }
#pragma unroll
    for (int fi = 0; fi < 4; ++fi) {
      int fb = f0 + fi * 16 + 8 * half;   // feature base (8 consecutive)
      float bb[8];
      *(float4*)&bb[0] = *(const float4*)(bias + fb);
      *(float4*)&bb[4] = *(const float4*)(bias + fb + 4);
      if (EPI == 0 || EPI == 1) {
        BF8 p8;
#pragma unroll
        for (int r = 0; r < 8; ++r) {
          float v = acc[fi][ti][r] + bb[r];
          if (EPI == 1) v = gelu_f(v);
          p8.e[r] = (bf16)v;
        }
        *(uint4*)(outB + t * Nout + fb) = p8.u;
      } else {
        float o[8];
        const float* rp = res + trow * Nout + fb;
        *(float4*)&o[0] = *(const float4*)rp;
        *(float4*)&o[4] = *(const float4*)(rp + 4);
#pragma unroll
        for (int r = 0; r < 8; ++r) o[r] += acc[fi][ti][r] + bb[r];
        float* op = outF + trow * Nout + fb;
        *(float4*)op       = *(float4*)&o[0];
        *(float4*)(op + 4) = *(float4*)&o[4];
      }
    }
  }
}

// ---------------------------------------------------------------------------
// Per-(window, head) attention.  One wave per block.
//   scoresT = K @ Q^T  (softmax over keys = per-lane reduce + 1 shfl_xor(16))
//   attn -> LDS row-major attn[n][m] via packed ds_store_b128,
//   outT  = vT @ attn^T via WMMA (ds_load_b128 fragment feeds),
//   lane's 8 accumulators = 8 consecutive head-dims -> vector global stores.
// ---------------------------------------------------------------------------
__global__ __launch_bounds__(32)
void attn_kernel(const bf16* __restrict__ qkv, const float* __restrict__ bm,
                 bf16* __restrict__ outp) {
  __shared__ __align__(16) bf16 s_attn[64 * 64];  // attn[n][m]
  __shared__ __align__(16) bf16 s_vT[32 * 64];    // vT[d][m]

  int lane = threadIdx.x & 31;
  int lr = lane & 15, half = lane >> 4;
  int w = blockIdx.x;
  int h = blockIdx.y;
  const bf16* base = qkv + (long)w * NTOKW * (3 * DIM_);

  // stage V transposed: coalesced 8B global loads, 4-way ds scatter
  for (int it = lane; it < NTOKW * 8; it += 32) {
    int m = it >> 3;
    int dc = (it & 7) * 4;
    BF4 r4;
    r4.u = *(const uint2*)(base + (long)m * (3 * DIM_) + 2 * DIM_ + h * HD_ + dc);
    s_vT[(dc + 0) * 64 + m] = r4.e[0];
    s_vT[(dc + 1) * 64 + m] = r4.e[1];
    s_vT[(dc + 2) * 64 + m] = r4.e[2];
    s_vT[(dc + 3) * 64 + m] = r4.e[3];
  }
  __syncthreads();

  // scoresT[m][n] = sum_d K[m][d] * Q[n][d]   (HD=32: single WMMA step)
  f32x8 acc[4][4] = {};
  {
    Frag af[4], bf[4];
#pragma unroll
    for (int mi = 0; mi < 4; ++mi) {
      const bf16* p = base + (long)(mi * 16 + lr) * (3 * DIM_) + DIM_ + h * HD_ + half * 8;
      af[mi].u[0] = *(const uint4*)p;
      af[mi].u[1] = *(const uint4*)(p + 16);
    }
#pragma unroll
    for (int ni = 0; ni < 4; ++ni) {
      const bf16* p = base + (long)(ni * 16 + lr) * (3 * DIM_) + h * HD_ + half * 8;
      bf[ni].u[0] = *(const uint4*)p;
      bf[ni].u[1] = *(const uint4*)(p + 16);
    }
#pragma unroll
    for (int mi = 0; mi < 4; ++mi)
#pragma unroll
      for (int ni = 0; ni < 4; ++ni)
        acc[mi][ni] = __builtin_amdgcn_wmma_f32_16x16x32_bf16(
            false, af[mi].v, false, bf[ni].v, (short)0, acc[mi][ni], false, false);
  }

  // scale + fused bias/mask table (vectorized float4 loads, lane-contiguous m)
  const float scale = 0.17677669529663687f;  // 1/sqrt(32)
  int wr = w >> 6, wc = w & 63;
  int cls = ((wr == 63) ? 2 : 0) | ((wc == 63) ? 1 : 0);
  const float* bmh = bm + (((long)cls * HEADS_ + h) << 12);
#pragma unroll
  for (int ni = 0; ni < 4; ++ni) {
    int n = ni * 16 + lr;                    // query index
#pragma unroll
    for (int mi = 0; mi < 4; ++mi) {
      float bv[8];
      const float* bp = bmh + n * 64 + mi * 16 + 8 * half;
      *(float4*)&bv[0] = *(const float4*)bp;
      *(float4*)&bv[4] = *(const float4*)(bp + 4);
#pragma unroll
      for (int r = 0; r < 8; ++r)
        acc[mi][ni][r] = acc[mi][ni][r] * scale + bv[r];
    }
  }

  // softmax over keys; packed bf16 rows into LDS (one b128 store per 8 keys)
#pragma unroll
  for (int ni = 0; ni < 4; ++ni) {
    float mx = -3.0e38f;
#pragma unroll
    for (int mi = 0; mi < 4; ++mi)
#pragma unroll
      for (int r = 0; r < 8; ++r) mx = fmaxf(mx, acc[mi][ni][r]);
    mx = fmaxf(mx, __shfl_xor(mx, 16, 32));
    float sum = 0.0f;
#pragma unroll
    for (int mi = 0; mi < 4; ++mi)
#pragma unroll
      for (int r = 0; r < 8; ++r) {
        float e = __expf(acc[mi][ni][r] - mx);
        acc[mi][ni][r] = e;
        sum += e;
      }
    sum += __shfl_xor(sum, 16, 32);
    float inv = 1.0f / sum;
    int n = ni * 16 + lr;
#pragma unroll
    for (int mi = 0; mi < 4; ++mi) {
      BF8 p8;
#pragma unroll
      for (int r = 0; r < 8; ++r) p8.e[r] = (bf16)(acc[mi][ni][r] * inv);
      *(uint4*)&s_attn[n * 64 + mi * 16 + 8 * half] = p8.u;
    }
  }
  __syncthreads();

  // outT[d][n] = sum_m vT[d][m] * attn[n][m]   (K=64 -> two WMMA steps)
  f32x8 oacc[2][4] = {};             // [dt][nt]
  for (int k0 = 0; k0 < 64; k0 += 32) {
    Frag av[2], ab[4];
#pragma unroll
    for (int dt = 0; dt < 2; ++dt) {
      const bf16* p = &s_vT[(dt * 16 + lr) * 64 + k0 + half * 8];
      av[dt].u[0] = *(const uint4*)p;
      av[dt].u[1] = *(const uint4*)(p + 16);
    }
#pragma unroll
    for (int nt = 0; nt < 4; ++nt) {
      const bf16* p = &s_attn[(nt * 16 + lr) * 64 + k0 + half * 8];
      ab[nt].u[0] = *(const uint4*)p;
      ab[nt].u[1] = *(const uint4*)(p + 16);
    }
#pragma unroll
    for (int dt = 0; dt < 2; ++dt)
#pragma unroll
      for (int nt = 0; nt < 4; ++nt)
        oacc[dt][nt] = __builtin_amdgcn_wmma_f32_16x16x32_bf16(
            false, av[dt].v, false, ab[nt].v, (short)0, oacc[dt][nt], false, false);
  }

#pragma unroll
  for (int nt = 0; nt < 4; ++nt) {
    int n = nt * 16 + lr;
    bf16* op = outp + ((long)w * NTOKW + n) * DIM_ + h * HD_;
#pragma unroll
    for (int dt = 0; dt < 2; ++dt) {
      int db = dt * 16 + 8 * half;   // 8 consecutive head-dims
      BF8 p8;
#pragma unroll
      for (int r = 0; r < 8; ++r) p8.e[r] = (bf16)oacc[dt][nt][r];
      *(uint4*)(op + db) = p8.u;
    }
  }
}

// ---------------------------------------------------------------------------
// Host-side orchestration
// ---------------------------------------------------------------------------
extern "C" void kernel_launch(void* const* d_in, const int* in_sizes, int n_in,
                              void* d_out, int out_size, void* d_ws, size_t ws_size,
                              hipStream_t stream) {
  const float* x      = (const float*)d_in[0];
  const float* ln1_w  = (const float*)d_in[1];
  const float* ln1_b  = (const float*)d_in[2];
  const float* qkv_w  = (const float*)d_in[3];
  const float* qkv_b  = (const float*)d_in[4];
  const float* rel_b  = (const float*)d_in[5];
  const float* proj_w = (const float*)d_in[6];
  const float* proj_b = (const float*)d_in[7];
  const float* ln2_w  = (const float*)d_in[8];
  const float* ln2_b  = (const float*)d_in[9];
  const float* mlp_w1 = (const float*)d_in[10];
  const float* mlp_b1 = (const float*)d_in[11];
  const float* mlp_w2 = (const float*)d_in[12];
  const float* mlp_b2 = (const float*)d_in[13];
  float* out = (float*)d_out;

  // workspace carve-up (256B-aligned slices)
  char* ws = (char*)d_ws;
  size_t off = 0;
  auto carve = [&](size_t bytes) -> char* {
    char* p = ws + off;
    off = (off + bytes + 255) & ~(size_t)255;
    return p;
  };
  bf16*  wq   = (bf16*)carve((size_t)3 * DIM_ * DIM_ * 2);   // 384x128
  bf16*  wp   = (bf16*)carve((size_t)DIM_ * DIM_ * 2);       // 128x128
  bf16*  w1   = (bf16*)carve((size_t)HID_ * DIM_ * 2);       // 512x128
  bf16*  w2   = (bf16*)carve((size_t)DIM_ * HID_ * 2);       // 128x512
  float* bm   = (float*)carve((size_t)4 * HEADS_ * 64 * 64 * 4); // bias+mask table
  bf16*  xw   = (bf16*)carve((size_t)NTOK * DIM_ * 2);       // windowed LN1 out
  bf16*  qkvb = (bf16*)carve((size_t)NTOK * 3 * DIM_ * 2);   // qkv activations
  bf16*  atno = (bf16*)carve((size_t)NTOK * DIM_ * 2);       // attention out
  float* x1   = (float*)carve((size_t)NTOK * DIM_ * 4);      // first residual
  bf16*  xn2  = (bf16*)carve((size_t)NTOK * DIM_ * 2);       // LN2 out
  bf16*  hbuf = (bf16*)carve((size_t)NTOK * HID_ * 2);       // MLP hidden
  (void)ws_size; (void)n_in; (void)in_sizes; (void)out_size;

  // weights -> bf16, fused bias/mask table
  f2bf_kernel<<<(3 * DIM_ * DIM_ + 255) / 256, 256, 0, stream>>>(qkv_w, wq, 3 * DIM_ * DIM_);
  f2bf_kernel<<<(DIM_ * DIM_ + 255) / 256, 256, 0, stream>>>(proj_w, wp, DIM_ * DIM_);
  f2bf_kernel<<<(HID_ * DIM_ + 255) / 256, 256, 0, stream>>>(mlp_w1, w1, HID_ * DIM_);
  f2bf_kernel<<<(DIM_ * HID_ + 255) / 256, 256, 0, stream>>>(mlp_w2, w2, DIM_ * HID_);
  bm_kernel<<<(4 * HEADS_ * 64 * 64) / 256, 256, 0, stream>>>(rel_b, bm);

  // LN1 + shift + window partition
  ln_kernel<1><<<NTOK / 8, 256, 0, stream>>>(x, ln1_w, ln1_b, xw);

  // QKV projection: [262144 x 128] @ [128 x 384]
  gemm_kernel<0><<<dim3(NTOK / 64, (3 * DIM_) / 64), 32, 0, stream>>>(
      xw, wq, qkv_b, DIM_, 3 * DIM_, qkvb, nullptr, nullptr);

  // window attention
  attn_kernel<<<dim3(NWIN, HEADS_), 32, 0, stream>>>(qkvb, bm, atno);

  // proj + reverse shift/partition scatter + skip: x1 = x + scatter(out @ Wp^T + b)
  gemm_kernel<3><<<dim3(NTOK / 64, DIM_ / 64), 32, 0, stream>>>(
      atno, wp, proj_b, DIM_, DIM_, nullptr, x1, x);

  // LN2
  ln_kernel<0><<<NTOK / 8, 256, 0, stream>>>(x1, ln2_w, ln2_b, xn2);

  // MLP up + GELU: [262144 x 128] @ [128 x 512]
  gemm_kernel<1><<<dim3(NTOK / 64, HID_ / 64), 32, 0, stream>>>(
      xn2, w1, mlp_b1, DIM_, HID_, hbuf, nullptr, nullptr);

  // MLP down + residual -> final output (fp32)
  gemm_kernel<2><<<dim3(NTOK / 64, DIM_ / 64), 32, 0, stream>>>(
      hbuf, w2, mlp_b2, HID_, DIM_, nullptr, out, x1);
}
